// DPModel_71674414236256
// MI455X (gfx1250) — matrix operand order repaired and checked by reading, not compiled, not run on hardware
//
#include <hip/hip_runtime.h>
#include <hip/hip_bf16.h>

typedef __attribute__((ext_vector_type(16))) _Float16 v16h;
typedef __attribute__((ext_vector_type(8)))  _Float16 v8h;
typedef __attribute__((ext_vector_type(8)))  float    v8f;

#define N_ATOMS    8192
#define M_NBR      128
#define TYPE_SPLIT 4096
#define RCUT_INV   (1.0f / 6.0f)
#define LOG2E2     2.8853900817779268f   // 2*log2(e)

// ---------------- d_in leaf indexing (setup_inputs insertion order) ---------
#define IN_EMB_W(i,j,l) (3 + (((i)*2+(j))*3 + (l))*2)
#define IN_EMB_B(i,j,l) (IN_EMB_W(i,j,l)+1)
#define IN_FIT_L0W(i) (27 + (i)*6)
#define IN_FIT_L0B(i) (28 + (i)*6)
#define IN_FIT_L1W(i) (29 + (i)*6)
#define IN_FIT_L1B(i) (30 + (i)*6)
#define IN_FIT_FW(i)  (31 + (i)*6)
#define IN_FIT_FB(i)  (32 + (i)*6)
#define IN_TBIAS  39
#define IN_EBIAS  40
#define IN_SRMEAN 41
#define IN_SRSTD  42

// tanh from pre-scaled argument t = 2*log2(e)*x : branchless, saturating.
__device__ __forceinline__ float tanh_raw(float t) {
  float e = __builtin_amdgcn_exp2f(t);
  return 1.0f - 2.0f * __builtin_amdgcn_rcpf(e + 1.0f);
}
// tanh(x + bias) where bK = bias*2*log2(e) is precomputed: one v_fma front-end.
__device__ __forceinline__ float tanh_biased(float x, float bK) {
  return tanh_raw(fmaf(x, LOG2E2, bK));
}

// ---------------- WMMA helpers (CDNA5 16x16x32 f16 -> f32) ------------------
__device__ __forceinline__ v8f wmma_f16(v16h a, v16h b, v8f c) {
  return __builtin_amdgcn_wmma_f32_16x16x32_f16(false, a, false, b, (short)0, c,
                                                false, false);
}

// A fragment (16x32 f16) from row-major [m][ld]; two contiguous b128 loads.
__device__ __forceinline__ v16h load_frag_a(const _Float16* src, int ld) {
  int lane = threadIdx.x & 31;
  int m  = lane & 15;
  int kb = (lane < 16) ? 0 : 16;
  const _Float16* row = src + m * ld + kb;
  v8h lo = *(const v8h*)(row);        // k = kb .. kb+7
  v8h hi = *(const v8h*)(row + 8);    // k = kb+8 .. kb+15
  return __builtin_shufflevector(lo, hi, 0, 1, 2, 3, 4, 5, 6, 7,
                                 8, 9, 10, 11, 12, 13, 14, 15);
}

// B fragment from 16 contiguous halves at p (caller computed lane offset).
__device__ __forceinline__ v16h load_b_contig(const _Float16* p) {
  v8h lo = *(const v8h*)(p);
  v8h hi = *(const v8h*)(p + 8);
  return __builtin_shufflevector(lo, hi, 0, 1, 2, 3, 4, 5, 6, 7,
                                 8, 9, 10, 11, 12, 13, 14, 15);
}

__device__ __forceinline__ v8f zero_c() {
  v8f z;
#pragma unroll
  for (int v = 0; v < 8; ++v) z[v] = 0.0f;
  return z;
}

// ---------------- kernel: init output with Ebias terms ----------------------
__global__ void k_init(const float* __restrict__ ebias, float* __restrict__ out) {
  if (threadIdx.x == 0 && blockIdx.x == 0)
    out[0] = 4096.0f * (ebias[0] + ebias[1]);
}

// ---------------- kernel: pack weights into B-fragment-linear f16 -----------
struct K0Params {
  const float* w0[2];        // fit (1024,128)
  const float* w1[2];        // fit (128,128)
  const float* ew1[2][2];    // embed layer1 (32,32)
  const float* ew2[2][2];    // embed layer2 (32,64)
  _Float16* w0p;             // [i][nt8][kc32][lane][16]   262144
  _Float16* w1p;             // [i][nt8][kc4][lane][16]     32768
  _Float16* ew1p;            // [i][j][nt2][lane][16]        4096
  _Float16* ew2p;            // [i][j][nt4][lane][16]        8192
  _Float16* idenp;           // [q2][lane][16]               1024
};

__global__ void k0_pack(K0Params p) {
  int idx = blockIdx.x * blockDim.x + threadIdx.x;
  if (idx < 262144) {
    int i = idx >> 17, r = idx & 131071;
    int e = r & 15, lane = (r >> 4) & 31, kc = (r >> 9) & 31, nt = (r >> 14) & 7;
    int n = nt * 16 + (lane & 15);
    int k = kc * 32 + ((lane < 16) ? 0 : 16) + e;
    p.w0p[idx] = (_Float16)p.w0[i][k * 128 + n];
  } else if (idx < 294912) {
    int r2 = idx - 262144;
    int i = r2 >> 14, r = r2 & 16383;
    int e = r & 15, lane = (r >> 4) & 31, kc = (r >> 9) & 3, nt = (r >> 11) & 7;
    int n = nt * 16 + (lane & 15);
    int k = kc * 32 + ((lane < 16) ? 0 : 16) + e;
    p.w1p[r2] = (_Float16)p.w1[i][k * 128 + n];
  } else if (idx < 299008) {
    int r3 = idx - 294912;
    int i = r3 >> 11, j = (r3 >> 10) & 1, r = r3 & 1023;
    int e = r & 15, lane = (r >> 4) & 31, nt = (r >> 9) & 1;
    int n = nt * 16 + (lane & 15);
    int k = ((lane < 16) ? 0 : 16) + e;
    p.ew1p[r3] = (_Float16)p.ew1[i][j][k * 32 + n];
  } else if (idx < 307200) {
    int r4 = idx - 299008;
    int i = r4 >> 12, j = (r4 >> 11) & 1, r = r4 & 2047;
    int e = r & 15, lane = (r >> 4) & 31, nt = (r >> 9) & 3;
    int n = nt * 16 + (lane & 15);
    int k = ((lane < 16) ? 0 : 16) + e;
    p.ew2p[r4] = (_Float16)p.ew2[i][j][k * 64 + n];
  } else if (idx < 308224) {
    // identity B fragments: I(32x32) column tiles q=0,1, fragment-linear
    int r5 = idx - 307200;
    int q = r5 >> 9, r = r5 & 511;
    int e = r & 15, lane = r >> 4;
    int n = q * 16 + (lane & 15);
    int k = ((lane < 16) ? 0 : 16) + e;
    p.idenp[r5] = (_Float16)((k == n) ? 1.0f : 0.0f);
  }
}

// ---------------- kernel 1: geometry + embed MLP + T + G --------------------
struct K1Params {
  const float* coord;
  const float* box;
  const int*   nbrs;
  const float* w0[2][2]; const float* b0[2][2];
  const float* b1[2][2]; const float* b2[2][2];
  const _Float16* ew1p;
  const _Float16* ew2p;
  const _Float16* idenp;
  const float* Tbias; const float* sr_mean; const float* sr_std;
  _Float16* gbuf;        // [N][1024] f16
};

__global__ __launch_bounds__(128) void k1_embed(K1Params p) {
  // weights/biases for layer0 and all biases pre-scaled by 2*log2(e)
  __shared__ float    sW0f[2][32], sB0f[2][32];
  __shared__ float    sB1f[2][32], sB2f[2][64];
  __shared__ alignas(16) _Float16 sW1p[2048];
  __shared__ alignas(16) _Float16 sW2p[4096];
  __shared__ alignas(16) _Float16 sH0[128][32];
  __shared__ alignas(16) _Float16 sH1[128][32];
  __shared__ alignas(16) _Float16 sEt[64][128];   // transposed E: [w][pair]
  __shared__ alignas(16) _Float16 sR[4][128];
  __shared__ float    sTpart[4][4][64];
  __shared__ float    sT[4][64];

  const int n    = blockIdx.x;
  const int tid  = threadIdx.x;
  const int lane = tid & 31;
  const int wave = tid >> 5;
  const int itype = (n >= TYPE_SPLIT) ? 1 : 0;

  const v16h idenB0 = load_b_contig(p.idenp + lane * 16);
  const v16h idenB1 = load_b_contig(p.idenp + 512 + lane * 16);

  // ---- stage packed embed weights into LDS ----
  {
    const uint4* s1 = (const uint4*)(p.ew1p + itype * 2048);
    uint4* d1 = (uint4*)sW1p;
    for (int idx = tid; idx < 256; idx += 128) d1[idx] = s1[idx];
    const uint4* s2 = (const uint4*)(p.ew2p + itype * 4096);
    uint4* d2 = (uint4*)sW2p;
    for (int idx = tid; idx < 512; idx += 128) d2[idx] = s2[idx];
  }
  if (tid < 32) {
    sW0f[0][tid] = p.w0[itype][0][tid] * LOG2E2;
    sW0f[1][tid] = p.w0[itype][1][tid] * LOG2E2;
    sB0f[0][tid] = p.b0[itype][0][tid] * LOG2E2;
    sB0f[1][tid] = p.b0[itype][1][tid] * LOG2E2;
    sB1f[0][tid] = p.b1[itype][0][tid] * LOG2E2;
    sB1f[1][tid] = p.b1[itype][1][tid] * LOG2E2;
  }
  if (tid < 64) {
    sB2f[0][tid] = p.b2[itype][0][tid] * LOG2E2;
    sB2f[1][tid] = p.b2[itype][1][tid] * LOG2E2;
  }
  __syncthreads();

  // ---- geometry + layer0 (one pair per thread) ----
  {
    const int m  = tid;
    const int nb = p.nbrs[n * M_NBR + m];
    const float Lx = p.box[0], Ly = p.box[4], Lz = p.box[8];
    float dx = p.coord[nb] - p.coord[n];
    float dy = p.coord[N_ATOMS + nb] - p.coord[N_ATOMS + n];
    float dz = p.coord[2 * N_ATOMS + nb] - p.coord[2 * N_ATOMS + n];
    dx -= Lx * rintf(dx / Lx);
    dy -= Ly * rintf(dy / Ly);
    dz -= Lz * rintf(dz / Lz);
    float r  = sqrtf(dx * dx + dy * dy + dz * dz + 1e-16f);
    float u  = r * RCUT_INV;
    float sw = 0.0f;
    if (u < 1.0f) sw = u * u * u * (-6.0f * u * u + 15.0f * u - 10.0f) + 1.0f;
    float sr = (r > 1e-4f) ? sw / fmaxf(r, 1e-4f) : 0.0f;
    float mean = p.sr_mean[itype], stdv = p.sr_std[itype];
    float scn = (sr - mean) / stdv;
    float srn = sr / stdv;
    float inv_r = 1.0f / (r + 1e-16f);
    float s3 = 1.7320508075688772f * srn * inv_r;
    sR[0][m] = (_Float16)srn;
    sR[1][m] = (_Float16)(s3 * dx);
    sR[2][m] = (_Float16)(s3 * dy);
    sR[3][m] = (_Float16)(s3 * dz);
    const int j = (m >= 64) ? 1 : 0;
#pragma unroll
    for (int c = 0; c < 32; ++c)
      sH0[m][c] = (_Float16)tanh_raw(fmaf(scn, sW0f[j][c], sB0f[j][c]));
  }
  __syncthreads();

  // ---- layer1 via WMMA; inline-0 C, bias folded into tanh fma ----
  for (int t = 2 * wave; t < 2 * wave + 2; ++t) {
    const int base = t * 16;
    const int j = (base >= 64) ? 1 : 0;
    v16h a = load_frag_a(&sH0[base][0], 32);
    v8f h0c[2];
    h0c[0] = wmma_f16(a, idenB0, zero_c());
    h0c[1] = wmma_f16(a, idenB1, zero_c());
#pragma unroll
    for (int ntile = 0; ntile < 2; ++ntile) {
      v16h b = load_b_contig(sW1p + j * 1024 + ntile * 512 + lane * 16);
      const int col = ntile * 16 + (lane & 15);
      const float bK = sB1f[j][col];
      v8f c = wmma_f16(a, b, zero_c());
#pragma unroll
      for (int v = 0; v < 8; ++v) {
        int mr = v + ((lane < 16) ? 0 : 8);
        sH1[base + mr][col] = (_Float16)(h0c[ntile][v] + tanh_biased(c[v], bK));
      }
    }
  }
  __syncthreads();

  // ---- layer2 via WMMA; concat-residual via identity-WMMA; E transposed ----
  for (int t = 2 * wave; t < 2 * wave + 2; ++t) {
    const int base = t * 16;
    const int j = (base >= 64) ? 1 : 0;
    v16h a = load_frag_a(&sH1[base][0], 32);
    v8f h1c[2];
    h1c[0] = wmma_f16(a, idenB0, zero_c());
    h1c[1] = wmma_f16(a, idenB1, zero_c());
#pragma unroll
    for (int ntile = 0; ntile < 4; ++ntile) {
      v16h b = load_b_contig(sW2p + j * 2048 + ntile * 512 + lane * 16);
      const int col = ntile * 16 + (lane & 15);
      const float bK = sB2f[j][col];
      v8f c = wmma_f16(a, b, zero_c());
#pragma unroll
      for (int v = 0; v < 8; ++v) {
        int mr = v + ((lane < 16) ? 0 : 8);
        sEt[col][base + mr] =
            (_Float16)(h1c[ntile & 1][v] + tanh_biased(c[v], bK));
      }
    }
  }
  __syncthreads();

  // ---- T partial: rows 0..3 = R, K = this wave's 32 pairs, via WMMA ----
  {
    const int koff = 32 * wave;
    const int m  = lane & 15;
    const int kb = (lane < 16) ? 0 : 16;
    v16h a;
    if (m < 4) {
      const _Float16* row = &sR[m][koff + kb];
      v8h lo = *(const v8h*)(row);
      v8h hi = *(const v8h*)(row + 8);
      a = __builtin_shufflevector(lo, hi, 0, 1, 2, 3, 4, 5, 6, 7,
                                  8, 9, 10, 11, 12, 13, 14, 15);
    } else {
#pragma unroll
      for (int e = 0; e < 16; ++e) a[e] = (_Float16)0.0f;
    }
#pragma unroll
    for (int ntile = 0; ntile < 4; ++ntile) {
      const int nw = ntile * 16 + (lane & 15);
      const int kbB = (lane < 16) ? 0 : 16;
      v16h b = load_b_contig(&sEt[nw][koff + kbB]);
      v8f c = wmma_f16(a, b, zero_c());
#pragma unroll
      for (int v = 0; v < 8; ++v) {
        int mr = v + ((lane < 16) ? 0 : 8);
        if (mr < 4) sTpart[wave][mr][ntile * 16 + (lane & 15)] = c[v];
      }
    }
  }
  __syncthreads();

  // ---- reduce partials, /NNBRS, add Tbias to row 0 ----
  if (tid < 64) {
#pragma unroll
    for (int x = 0; x < 4; ++x) {
      float s = sTpart[0][x][tid] + sTpart[1][x][tid] +
                sTpart[2][x][tid] + sTpart[3][x][tid];
      s *= (1.0f / 64.0f);
      if (x == 0) s += p.Tbias[tid];
      sT[x][tid] = s;
    }
  }
  __syncthreads();

  // ---- G[a][w] = sum_x T[x][w]*T[x][a]; vectorized b128 store ----
  {
    const int a = tid >> 3;
    const int wbase = (tid & 7) * 8;
    float t0a = sT[0][a], t1a = sT[1][a], t2a = sT[2][a], t3a = sT[3][a];
    v8h g8;
#pragma unroll
    for (int q = 0; q < 8; ++q) {
      int w = wbase + q;
      float g = sT[0][w] * t0a + sT[1][w] * t1a +
                sT[2][w] * t2a + sT[3][w] * t3a;
      g8[q] = (_Float16)g;
    }
    *(v8h*)(p.gbuf + (size_t)n * 1024 + tid * 8) = g8;
  }
}

// ---------------- kernel 2: fit network + reduction -------------------------
struct K2Params {
  const _Float16* gbuf;
  const _Float16* w0p;
  const _Float16* w1p;
  const _Float16* idenp;
  const float* b0[2]; const float* b1[2];
  const float* wf[2]; const float* bf[2];
  float* out;
};

__global__ __launch_bounds__(256) void k2_fit(K2Params p) {
  __shared__ alignas(16) _Float16 sG[16 * 1024];
  __shared__ alignas(16) _Float16 sH[16][128];
  __shared__ float    sH2[16][128];
  __shared__ float    sPart[16][16];
  __shared__ float    sEsum[16];

  const int tid  = threadIdx.x;
  const int lane = tid & 31;
  const int ntile = tid >> 5;  // 8 waves -> 8 column tiles of 16
  const int base = blockIdx.x * 16;
  const int i = (base >= TYPE_SPLIT) ? 1 : 0;

  const v16h idenQ = load_b_contig(p.idenp + (ntile & 1) * 512 + lane * 16);

  // stage 16 atoms' g vectors (32 KB) into LDS
  {
    const uint4* src = (const uint4*)(p.gbuf + (size_t)base * 1024);
    uint4* dst = (uint4*)sG;
    for (int idx = tid; idx < 2048; idx += 256) dst[idx] = src[idx];
  }
  __syncthreads();

  const int col = ntile * 16 + (lane & 15);
  const float b0K = p.b0[i][col] * LOG2E2;
  const float b1K = p.b1[i][col] * LOG2E2;

  // layer0: h = tanh(g @ W0 + b0), K = 1024; inline-0 C, bias in tanh fma
  v8f acc = zero_c();
  for (int kc = 0; kc < 32; ++kc) {
    v16h a = load_frag_a(sG + kc * 32, 1024);
    v16h b = load_b_contig(p.w0p +
        ((((size_t)i * 8 + ntile) * 32 + kc) * 32 + lane) * 16);
    acc = wmma_f16(a, b, acc);
  }
#pragma unroll
  for (int v = 0; v < 8; ++v) {
    int mr = v + ((lane < 16) ? 0 : 8);
    sH[mr][col] = (_Float16)tanh_biased(acc[v], b0K);
  }
  __syncthreads();

  // layer1: h2 = h + tanh(h @ W1 + b1); residual via identity-WMMA
  v8f acc2 = zero_c();
  v16h aSave;
#pragma unroll
  for (int kc = 0; kc < 4; ++kc) {
    v16h a = load_frag_a(&sH[0][0] + kc * 32, 128);
    if (kc == (ntile >> 1)) aSave = a;
    v16h b = load_b_contig(p.w1p +
        ((((size_t)i * 8 + ntile) * 4 + kc) * 32 + lane) * 16);
    acc2 = wmma_f16(a, b, acc2);
  }
  {
    v8f hres = wmma_f16(aSave, idenQ, zero_c());
#pragma unroll
    for (int v = 0; v < 8; ++v) {
      int mr = v + ((lane < 16) ? 0 : 8);
      sH2[mr][col] = hres[v] + tanh_biased(acc2[v], b1K);
    }
  }
  __syncthreads();

  // final: parallel dot with wf, tree reduce, atomicAdd
  {
    const int aat = tid >> 4;       // atom 0..15
    const int seg = tid & 15;       // 8-col segment
    const float* wf = p.wf[i];
    float partial = 0.0f;
#pragma unroll
    for (int q = 0; q < 8; ++q) {
      int c = seg * 8 + q;
      partial += sH2[aat][c] * wf[c];
    }
    sPart[aat][seg] = partial;
  }
  __syncthreads();
  if (tid < 16) {
    float dot = 0.0f;
#pragma unroll
    for (int s = 0; s < 16; ++s) dot += sPart[tid][s];
    sEsum[tid] = dot + p.bf[i][0];
  }
  __syncthreads();
  if (tid == 0) {
    float s = 0.0f;
#pragma unroll
    for (int k = 0; k < 16; ++k) s += sEsum[k];
    atomicAdd(p.out, s);
  }
}

// ---------------- host launcher ---------------------------------------------
extern "C" void kernel_launch(void* const* d_in, const int* in_sizes, int n_in,
                              void* d_out, int out_size, void* d_ws, size_t ws_size,
                              hipStream_t stream) {
  (void)in_sizes; (void)n_in; (void)out_size; (void)ws_size;

  char* ws = (char*)d_ws;
  _Float16* gbuf  = (_Float16*)ws;                                  // 16 MB
  _Float16* w0p   = (_Float16*)(ws + (size_t)N_ATOMS * 1024 * 2);   // 512 KB
  _Float16* w1p   = w0p + 262144;                                   // 64 KB
  _Float16* ew1p  = w1p + 32768;                                    // 8 KB
  _Float16* ew2p  = ew1p + 4096;                                    // 16 KB
  _Float16* idenp = ew2p + 8192;                                    // 2 KB

  k_init<<<1, 32, 0, stream>>>((const float*)d_in[IN_EBIAS], (float*)d_out);

  K0Params p0;
  for (int i = 0; i < 2; ++i) {
    p0.w0[i] = (const float*)d_in[IN_FIT_L0W(i)];
    p0.w1[i] = (const float*)d_in[IN_FIT_L1W(i)];
    for (int j = 0; j < 2; ++j) {
      p0.ew1[i][j] = (const float*)d_in[IN_EMB_W(i, j, 1)];
      p0.ew2[i][j] = (const float*)d_in[IN_EMB_W(i, j, 2)];
    }
  }
  p0.w0p = w0p; p0.w1p = w1p; p0.ew1p = ew1p; p0.ew2p = ew2p; p0.idenp = idenp;
  k0_pack<<<1204, 256, 0, stream>>>(p0);

  K1Params p1;
  p1.coord = (const float*)d_in[0];
  p1.box   = (const float*)d_in[1];
  p1.nbrs  = (const int*)d_in[2];
  for (int i = 0; i < 2; ++i)
    for (int j = 0; j < 2; ++j) {
      p1.w0[i][j] = (const float*)d_in[IN_EMB_W(i, j, 0)];
      p1.b0[i][j] = (const float*)d_in[IN_EMB_B(i, j, 0)];
      p1.b1[i][j] = (const float*)d_in[IN_EMB_B(i, j, 1)];
      p1.b2[i][j] = (const float*)d_in[IN_EMB_B(i, j, 2)];
    }
  p1.ew1p    = ew1p;
  p1.ew2p    = ew2p;
  p1.idenp   = idenp;
  p1.Tbias   = (const float*)d_in[IN_TBIAS];
  p1.sr_mean = (const float*)d_in[IN_SRMEAN];
  p1.sr_std  = (const float*)d_in[IN_SRSTD];
  p1.gbuf    = gbuf;
  k1_embed<<<N_ATOMS, 128, 0, stream>>>(p1);

  K2Params p2;
  p2.gbuf = gbuf; p2.w0p = w0p; p2.w1p = w1p; p2.idenp = idenp;
  for (int i = 0; i < 2; ++i) {
    p2.b0[i] = (const float*)d_in[IN_FIT_L0B(i)];
    p2.b1[i] = (const float*)d_in[IN_FIT_L1B(i)];
    p2.wf[i] = (const float*)d_in[IN_FIT_FW(i)];
    p2.bf[i] = (const float*)d_in[IN_FIT_FB(i)];
  }
  p2.out = (float*)d_out;
  k2_fit<<<N_ATOMS / 16, 256, 0, stream>>>(p2);
}
